// InterpolatedLinear_53283364274742
// MI455X (gfx1250) — compile-verified, matrix operand order checked by reading, not run on hardware
//
#include <hip/hip_runtime.h>

typedef __attribute__((ext_vector_type(16))) __bf16        v16bf;
typedef __attribute__((ext_vector_type(8)))  float         v8f;
typedef __attribute__((ext_vector_type(4)))  unsigned int  u32x4;
typedef __attribute__((ext_vector_type(4)))  int           v4i;

#define IN_F   2048
#define OUT_F  2048
#define BATCH  8192

#define BM 128   // block tile M (batch rows)
#define BN 64    // block tile N (output features)
#define BK 64    // K per stage (2 wmma k-steps of 32)
#define NKB (IN_F / BK)   // 32 stages

union FragU { v16bf v; u32x4 q[2]; };

// ---- CDNA5 async global->LDS copy path (ASYNCcnt), if the toolchain has it --
#if defined(__has_builtin)
#if __has_builtin(__builtin_amdgcn_global_load_async_to_lds_b128)
#define HAVE_ASYNC_LDS 1
#endif
#endif

#define AS_GLOBAL __attribute__((address_space(1)))
#define AS_LDS    __attribute__((address_space(3)))

#ifdef HAVE_ASYNC_LDS
__device__ __forceinline__ void async_cp16(const void* g, void* l) {
  __builtin_amdgcn_global_load_async_to_lds_b128(
      (AS_GLOBAL v4i*)(g), (AS_LDS v4i*)(l), 0, 0);
}
#endif

__device__ __forceinline__ void wait_async0() {
#if defined(__has_builtin) && __has_builtin(__builtin_amdgcn_s_wait_asynccnt)
  __builtin_amdgcn_s_wait_asynccnt(0);
#else
  asm volatile("s_wait_asynccnt 0x0" ::: "memory");
#endif
}

// ---------------------------------------------------------------------------
// Pass 1: one-shot f32 -> bf16 conversion (native casts -> hw cvt path)
// ---------------------------------------------------------------------------
__global__ __launch_bounds__(256)
void cvt_f32_to_bf16(const float* __restrict__ src, __bf16* __restrict__ dst, int n)
{
  int i = (blockIdx.x * blockDim.x + threadIdx.x) * 8;
  const int stride = gridDim.x * blockDim.x * 8;
  for (; i < n; i += stride) {
    float4 a = *(const float4*)(src + i);
    float4 b = *(const float4*)(src + i + 4);
    union { __bf16 h[8]; u32x4 q; } u;
    u.h[0] = (__bf16)a.x; u.h[1] = (__bf16)a.y;
    u.h[2] = (__bf16)a.z; u.h[3] = (__bf16)a.w;
    u.h[4] = (__bf16)b.x; u.h[5] = (__bf16)b.y;
    u.h[6] = (__bf16)b.z; u.h[7] = (__bf16)b.w;
    *(u32x4*)(dst + i) = u.q;
  }
}

// ---------------------------------------------------------------------------
// Pass 2: fused dual-GEMM + interpolation, bf16 inputs
// ---------------------------------------------------------------------------
__global__ __launch_bounds__(256)
void interp_gemm_bf16(const __bf16* __restrict__ Xb,
                      const __bf16* __restrict__ W1b,
                      const __bf16* __restrict__ W2b,
                      const float* __restrict__ alpha,
                      const float* __restrict__ b1,
                      const float* __restrict__ b2,
                      float* __restrict__ Y)
{
  __shared__ unsigned int sX [2][BM * (BK / 2)];  // 16 KB / stage
  __shared__ unsigned int sW1[2][BN * (BK / 2)];  //  8 KB / stage
  __shared__ unsigned int sW2[2][BN * (BK / 2)];  //  8 KB / stage

  const int tid  = threadIdx.x;
  const int lane = tid & 31;
  const int wid  = tid >> 5;
  const int wm   = wid & 3;    // 4 waves along M
  const int wn   = wid >> 2;   // 2 waves along N
  const int l15  = lane & 15;
  const int lhi  = lane >> 4;

  const int m0 = blockIdx.y * BM;
  const int n0 = blockIdx.x * BN;

  v8f acc1[2][2], acc2[2][2];
#pragma unroll
  for (int tm = 0; tm < 2; ++tm)
#pragma unroll
    for (int tn = 0; tn < 2; ++tn) {
      acc1[tm][tn] = (v8f){0.f,0.f,0.f,0.f,0.f,0.f,0.f,0.f};
      acc2[tm][tn] = (v8f){0.f,0.f,0.f,0.f,0.f,0.f,0.f,0.f};
    }

#ifdef HAVE_ASYNC_LDS
  // ---- async tile mover: global -> LDS directly, tracked by ASYNCcnt ----
  auto gissue = [&](int kb, int s) {
    const int k0 = kb * BK;
#pragma unroll
    for (int i = 0; i < 4; ++i) {           // X: 128 rows x 8 chunks of 8 bf16
      int idx = tid + i * 256;
      int row = idx >> 3, c = idx & 7;
      async_cp16(Xb + (size_t)(m0 + row) * IN_F + k0 + c * 8,
                 &sX[s][row * (BK / 2) + c * 4]);
    }
#pragma unroll
    for (int i = 0; i < 2; ++i) {           // W: 64 rows x 8 chunks
      int idx = tid + i * 256;
      int row = idx >> 3, c = idx & 7;
      async_cp16(W1b + (size_t)(n0 + row) * IN_F + k0 + c * 8,
                 &sW1[s][row * (BK / 2) + c * 4]);
      async_cp16(W2b + (size_t)(n0 + row) * IN_F + k0 + c * 8,
                 &sW2[s][row * (BK / 2) + c * 4]);
    }
  };
#else
  u32x4 rx[4], rw1[2], rw2[2];              // register-staged fallback
  auto gload = [&](int kb) {
    const int k0 = kb * BK;
#pragma unroll
    for (int i = 0; i < 4; ++i) {
      int idx = tid + i * 256;
      int row = idx >> 3, c = idx & 7;
      rx[i] = *(const u32x4*)(Xb + (size_t)(m0 + row) * IN_F + k0 + c * 8);
    }
#pragma unroll
    for (int i = 0; i < 2; ++i) {
      int idx = tid + i * 256;
      int row = idx >> 3, c = idx & 7;
      rw1[i] = *(const u32x4*)(W1b + (size_t)(n0 + row) * IN_F + k0 + c * 8);
      rw2[i] = *(const u32x4*)(W2b + (size_t)(n0 + row) * IN_F + k0 + c * 8);
    }
  };
  auto sstore = [&](int s) {
#pragma unroll
    for (int i = 0; i < 4; ++i) {
      int idx = tid + i * 256;
      int row = idx >> 3, c = idx & 7;
      *(u32x4*)&sX[s][row * (BK / 2) + c * 4] = rx[i];
    }
#pragma unroll
    for (int i = 0; i < 2; ++i) {
      int idx = tid + i * 256;
      int row = idx >> 3, c = idx & 7;
      *(u32x4*)&sW1[s][row * (BK / 2) + c * 4] = rw1[i];
      *(u32x4*)&sW2[s][row * (BK / 2) + c * 4] = rw2[i];
    }
  };
#endif

  auto compute = [&](int s) {
#pragma unroll
    for (int ko = 0; ko < BK; ko += 32) {
      const int kdw = (ko + lhi * 8) >> 1;      // lanes16-31 take the +8 K halves
      FragU a[2], bwa[2], bwb[2];
#pragma unroll
      for (int tm = 0; tm < 2; ++tm) {
        int row = wm * 32 + tm * 16 + l15;
        const unsigned int* p = &sX[s][row * (BK / 2) + kdw];
        a[tm].q[0] = *(const u32x4*)(p);        // K = kbase .. kbase+7
        a[tm].q[1] = *(const u32x4*)(p + 8);    // K = kbase+16 .. kbase+23
      }
#pragma unroll
      for (int tn = 0; tn < 2; ++tn) {
        int col = wn * 32 + tn * 16 + l15;
        const unsigned int* p1 = &sW1[s][col * (BK / 2) + kdw];
        bwa[tn].q[0] = *(const u32x4*)(p1);
        bwa[tn].q[1] = *(const u32x4*)(p1 + 8);
        const unsigned int* p2 = &sW2[s][col * (BK / 2) + kdw];
        bwb[tn].q[0] = *(const u32x4*)(p2);
        bwb[tn].q[1] = *(const u32x4*)(p2 + 8);
      }
#pragma unroll
      for (int tm = 0; tm < 2; ++tm)
#pragma unroll
        for (int tn = 0; tn < 2; ++tn) {
          acc1[tm][tn] = __builtin_amdgcn_wmma_f32_16x16x32_bf16(
              false, a[tm].v, false, bwa[tn].v, (short)0, acc1[tm][tn], false, false);
          acc2[tm][tn] = __builtin_amdgcn_wmma_f32_16x16x32_bf16(
              false, a[tm].v, false, bwb[tn].v, (short)0, acc2[tm][tn], false, false);
        }
    }
  };

#ifdef HAVE_ASYNC_LDS
  gissue(0, 0);
  wait_async0();
  __syncthreads();
  for (int kb = 0; kb < NKB; ++kb) {
    const int s = kb & 1;
    if (kb + 1 < NKB) gissue(kb + 1, s ^ 1);  // overlaps with compute below
    compute(s);
    if (kb + 1 < NKB) {
      wait_async0();
      __syncthreads();
    }
  }
#else
  gload(0);
  sstore(0);
  __syncthreads();
  for (int kb = 0; kb < NKB; ++kb) {
    const int s = kb & 1;
    if (kb + 1 < NKB) gload(kb + 1);
    compute(s);
    if (kb + 1 < NKB) {
      sstore(s ^ 1);
      __syncthreads();
    }
  }
#endif

  // epilogue: y = a*(y1+b1) + (1-a)*(y2+b2)
  const int mb = m0 + wm * 32;
  const int nb = n0 + wn * 32;
#pragma unroll
  for (int tm = 0; tm < 2; ++tm) {
    float av[8];
#pragma unroll
    for (int r = 0; r < 8; ++r)
      av[r] = alpha[mb + tm * 16 + lhi * 8 + r];
#pragma unroll
    for (int tn = 0; tn < 2; ++tn) {
      const int gn = nb + tn * 16 + l15;
      const float b1v = b1[gn];
      const float b2v = b2[gn];
#pragma unroll
      for (int r = 0; r < 8; ++r) {
        const int gm = mb + tm * 16 + lhi * 8 + r;
        const float a = av[r];
        Y[(size_t)gm * OUT_F + gn] =
            a * (acc1[tm][tn][r] + b1v) + (1.0f - a) * (acc2[tm][tn][r] + b2v);
      }
    }
  }
}

// ---------------------------------------------------------------------------
// Fallback: fused-conversion kernel (used only if ws is too small)
// ---------------------------------------------------------------------------
__device__ __forceinline__ unsigned int pack2_bf16(float a, float b) {
  union { float f; unsigned int u; } ua, ub;
  ua.f = a; ub.f = b;
  unsigned int ra = (ua.u + 0x7FFFu + ((ua.u >> 16) & 1u)) >> 16;
  unsigned int rb = (ub.u + 0x7FFFu + ((ub.u >> 16) & 1u)) >> 16;
  return (ra & 0xFFFFu) | (rb << 16);
}

__global__ __launch_bounds__(256)
void interp_gemm_fused_cvt(const float* __restrict__ X,
                           const float* __restrict__ alpha,
                           const float* __restrict__ W1,
                           const float* __restrict__ W2,
                           const float* __restrict__ b1,
                           const float* __restrict__ b2,
                           float* __restrict__ Y)
{
  __shared__ unsigned int sX [2][BM * (BK / 2)];
  __shared__ unsigned int sW1[2][BN * (BK / 2)];
  __shared__ unsigned int sW2[2][BN * (BK / 2)];

  const int tid  = threadIdx.x;
  const int lane = tid & 31;
  const int wid  = tid >> 5;
  const int wm   = wid & 3;
  const int wn   = wid >> 2;
  const int l15  = lane & 15;
  const int lhi  = lane >> 4;

  const int m0 = blockIdx.y * BM;
  const int n0 = blockIdx.x * BN;

  v8f acc1[2][2], acc2[2][2];
#pragma unroll
  for (int tm = 0; tm < 2; ++tm)
#pragma unroll
    for (int tn = 0; tn < 2; ++tn) {
      acc1[tm][tn] = (v8f){0.f,0.f,0.f,0.f,0.f,0.f,0.f,0.f};
      acc2[tm][tn] = (v8f){0.f,0.f,0.f,0.f,0.f,0.f,0.f,0.f};
    }

  float4 rx[8], rw1[4], rw2[4];

  auto gload = [&](int kb) {
    const int k0 = kb * BK;
#pragma unroll
    for (int i = 0; i < 8; ++i) {
      int idx = tid + i * 256;
      int row = idx >> 4, c4 = idx & 15;
      rx[i] = *(const float4*)(X + (size_t)(m0 + row) * IN_F + k0 + c4 * 4);
    }
#pragma unroll
    for (int i = 0; i < 4; ++i) {
      int idx = tid + i * 256;
      int row = idx >> 4, c4 = idx & 15;
      rw1[i] = *(const float4*)(W1 + (size_t)(n0 + row) * IN_F + k0 + c4 * 4);
      rw2[i] = *(const float4*)(W2 + (size_t)(n0 + row) * IN_F + k0 + c4 * 4);
    }
  };

  auto sstore = [&](int s) {
#pragma unroll
    for (int i = 0; i < 8; ++i) {
      int idx = tid + i * 256;
      int row = idx >> 4, c4 = idx & 15;
      unsigned int* p = &sX[s][row * (BK / 2) + c4 * 2];
      p[0] = pack2_bf16(rx[i].x, rx[i].y);
      p[1] = pack2_bf16(rx[i].z, rx[i].w);
    }
#pragma unroll
    for (int i = 0; i < 4; ++i) {
      int idx = tid + i * 256;
      int row = idx >> 4, c4 = idx & 15;
      unsigned int* p1 = &sW1[s][row * (BK / 2) + c4 * 2];
      p1[0] = pack2_bf16(rw1[i].x, rw1[i].y);
      p1[1] = pack2_bf16(rw1[i].z, rw1[i].w);
      unsigned int* p2 = &sW2[s][row * (BK / 2) + c4 * 2];
      p2[0] = pack2_bf16(rw2[i].x, rw2[i].y);
      p2[1] = pack2_bf16(rw2[i].z, rw2[i].w);
    }
  };

  auto compute = [&](int s) {
#pragma unroll
    for (int ko = 0; ko < BK; ko += 32) {
      const int kdw = (ko + lhi * 8) >> 1;
      FragU a[2], bwa[2], bwb[2];
#pragma unroll
      for (int tm = 0; tm < 2; ++tm) {
        int row = wm * 32 + tm * 16 + l15;
        const unsigned int* p = &sX[s][row * (BK / 2) + kdw];
        a[tm].q[0] = *(const u32x4*)(p);
        a[tm].q[1] = *(const u32x4*)(p + 8);
      }
#pragma unroll
      for (int tn = 0; tn < 2; ++tn) {
        int col = wn * 32 + tn * 16 + l15;
        const unsigned int* p1 = &sW1[s][col * (BK / 2) + kdw];
        bwa[tn].q[0] = *(const u32x4*)(p1);
        bwa[tn].q[1] = *(const u32x4*)(p1 + 8);
        const unsigned int* p2 = &sW2[s][col * (BK / 2) + kdw];
        bwb[tn].q[0] = *(const u32x4*)(p2);
        bwb[tn].q[1] = *(const u32x4*)(p2 + 8);
      }
#pragma unroll
      for (int tm = 0; tm < 2; ++tm)
#pragma unroll
        for (int tn = 0; tn < 2; ++tn) {
          acc1[tm][tn] = __builtin_amdgcn_wmma_f32_16x16x32_bf16(
              false, a[tm].v, false, bwa[tn].v, (short)0, acc1[tm][tn], false, false);
          acc2[tm][tn] = __builtin_amdgcn_wmma_f32_16x16x32_bf16(
              false, a[tm].v, false, bwb[tn].v, (short)0, acc2[tm][tn], false, false);
        }
    }
  };

  gload(0);
  sstore(0);
  __syncthreads();
  for (int kb = 0; kb < NKB; ++kb) {
    const int s = kb & 1;
    if (kb + 1 < NKB) gload(kb + 1);
    compute(s);
    if (kb + 1 < NKB) {
      sstore(s ^ 1);
      __syncthreads();
    }
  }

  const int mb = m0 + wm * 32;
  const int nb = n0 + wn * 32;
#pragma unroll
  for (int tm = 0; tm < 2; ++tm) {
    float av[8];
#pragma unroll
    for (int r = 0; r < 8; ++r)
      av[r] = alpha[mb + tm * 16 + lhi * 8 + r];
#pragma unroll
    for (int tn = 0; tn < 2; ++tn) {
      const int gn = nb + tn * 16 + l15;
      const float b1v = b1[gn];
      const float b2v = b2[gn];
#pragma unroll
      for (int r = 0; r < 8; ++r) {
        const int gm = mb + tm * 16 + lhi * 8 + r;
        const float a = av[r];
        Y[(size_t)gm * OUT_F + gn] =
            a * (acc1[tm][tn][r] + b1v) + (1.0f - a) * (acc2[tm][tn][r] + b2v);
      }
    }
  }
}

// ---------------------------------------------------------------------------
extern "C" void kernel_launch(void* const* d_in, const int* in_sizes, int n_in,
                              void* d_out, int out_size, void* d_ws, size_t ws_size,
                              hipStream_t stream) {
  const float* X  = (const float*)d_in[0];  // [8192,2048]
  const float* al = (const float*)d_in[1];  // [8192]
  const float* W1 = (const float*)d_in[2];  // [2048,2048]
  const float* W2 = (const float*)d_in[3];  // [2048,2048]
  const float* b1 = (const float*)d_in[4];  // [2048]
  const float* b2 = (const float*)d_in[5];  // [2048]
  float* Y = (float*)d_out;                 // [8192,2048]

  const size_t nX = (size_t)BATCH * IN_F;       // 16,777,216
  const size_t nW = (size_t)OUT_F * IN_F;       //  4,194,304
  const size_t ws_needed = (nX + 2 * nW) * sizeof(__bf16);  // ~50.3 MB

  dim3 grid(OUT_F / BN, BATCH / BM);            // (32, 64)
  dim3 block(256);                              // 8 wave32s

  if (ws_size >= ws_needed) {
    __bf16* Xb  = (__bf16*)d_ws;
    __bf16* W1b = Xb + nX;
    __bf16* W2b = W1b + nW;
    cvt_f32_to_bf16<<<(int)(nX / 8 / 256), 256, 0, stream>>>(X,  Xb,  (int)nX);
    cvt_f32_to_bf16<<<(int)(nW / 8 / 256), 256, 0, stream>>>(W1, W1b, (int)nW);
    cvt_f32_to_bf16<<<(int)(nW / 8 / 256), 256, 0, stream>>>(W2, W2b, (int)nW);
    interp_gemm_bf16<<<grid, block, 0, stream>>>(Xb, W1b, W2b, al, b1, b2, Y);
  } else {
    interp_gemm_fused_cvt<<<grid, block, 0, stream>>>(X, al, W1, W2, b1, b2, Y);
  }
}
